// Fusion_27041114096196
// MI455X (gfx1250) — matrix operand order repaired and checked by reading, not compiled, not run on hardware
//
#include <hip/hip_runtime.h>
#include <hip/hip_bf16.h>
#include <stdint.h>

#define DEV static __device__ __forceinline__

typedef __attribute__((ext_vector_type(16))) __bf16   bf16x16;
typedef __attribute__((ext_vector_type(8)))  __bf16   bf16x8;
typedef __attribute__((ext_vector_type(8)))  float    f32x8;
typedef __attribute__((ext_vector_type(4)))  uint32_t u32x4;
typedef __attribute__((ext_vector_type(8)))  int      i32x8;
typedef __attribute__((ext_vector_type(4)))  int      i32x4;

#if defined(__gfx1250__) && __has_builtin(__builtin_amdgcn_tensor_load_to_lds) && \
    __has_builtin(__builtin_amdgcn_s_wait_tensorcnt)
#define USE_TDM 1
#else
#define USE_TDM 0
#endif

DEV __bf16 f2bf(float f) {
  union { float f; uint32_t u; } x; x.f = f;
  uint32_t r = (x.u + 0x7FFFu + ((x.u >> 16) & 1u)) >> 16;  // RNE
  union { uint16_t s; __bf16 b; } y; y.s = (uint16_t)r;
  return y.b;
}

DEV int imin(int a, int b) { return a < b ? a : b; }

// 0=none 1=relu 2=silu 3=softplus
template<int ACT> DEV float act_apply(float v) {
  if (ACT == 1) return v > 0.f ? v : 0.f;
  if (ACT == 2) return v / (1.f + __expf(-v));
  if (ACT == 3) return (v > 20.f) ? v : log1pf(__expf(v));
  return v;
}

// 16-element bf16 fragment from a Kp-strided row (16B-aligned halves -> b128)
DEV bf16x16 ldfrag(const __bf16* __restrict__ row, int kb, int hi) {
  union { bf16x16 v; bf16x8 h[2]; } u;
  u.h[0] = *(const bf16x8*)(row + kb + hi);
  u.h[1] = *(const bf16x8*)(row + kb + hi + 16);
  return u.v;
}

// fragment from an LDS-staged 64x32 B tile (row stride 32 elems = 64B)
DEV bf16x16 ldfrag_lds(const __bf16* base, int hi) {
  union { bf16x16 v; bf16x8 h[2]; } u;
  u.h[0] = *(const bf16x8*)(base + hi);
  u.h[1] = *(const bf16x8*)(base + hi + 16);
  return u.v;
}

#if USE_TDM
// TDM descriptor DMA: 64 rows x 32 bf16 from a (rows, Kp)-strided matrix tile
// into LDS at byte offset lds_off (row-major, contiguous 64B rows).
// D# group0: count=1 | lds_addr | global_addr | type=2.   [ISA 8.3]
// D# group1: data_size=2B, tensor_dim0=Kp, tensor_dim1=64, tile_dim0=32,
//            tile_dim1=64, tensor_dim0_stride=Kp.          [ISA 8.4]
// This toolchain's builtin takes 6 args: (g0, g1, g2, g3, g4, cpol).
DEV void tdm_load_b_tile(const __bf16* gsrc, uint32_t lds_off, int Kp) {
  const uint64_t ga = (uint64_t)(uintptr_t)gsrc;
  u32x4 g0;
  g0[0] = 1u;                                   // count=1, user mode, no gather
  g0[1] = lds_off;                              // lds_addr (bytes)
  g0[2] = (uint32_t)ga;                         // global_addr[31:0]
  g0[3] = (uint32_t)(ga >> 32) | (2u << 30);    // global_addr[56:32] | type=2
  i32x8 g1;
  g1[0] = 0x00010000;                           // data_size=1 (2 bytes)
  g1[1] = (int)(((uint32_t)Kp & 0xFFFFu) << 16);// tensor_dim0[15:0]
  g1[2] = (int)(64u << 16);                     // tensor_dim0[31:16]=0 | tensor_dim1[15:0]=64
  g1[3] = (int)(32u << 16);                     // tensor_dim1[31:16]=0 | tile_dim0=32
  g1[4] = 64;                                   // tile_dim1=64, tile_dim2=0
  g1[5] = Kp;                                   // tensor_dim0_stride[31:0]
  g1[6] = 0;                                    // stride[47:32]=0, dim1_stride lo=0
  g1[7] = 0;
  const i32x4 z4 = {0, 0, 0, 0};
  const i32x8 z8 = {0, 0, 0, 0, 0, 0, 0, 0};
  __builtin_amdgcn_tensor_load_to_lds(g0, g1, z4, z4, z8, 0);
}
#endif

// ---------------------------------------------------------------------------
// bf16 GEMM: C[M,N] = act( A[M,Kp] @ W[N,Kp]^T + bias ), K pre-padded w/ zeros.
// Block = 4 waves covering a 64(M) x 64(N) C tile.  The 64x32 B tile is staged
// into LDS with double buffering: via the Tensor Data Mover (TENSORcnt) when
// available, else cooperatively through VGPRs.  The next A fragment is
// prefetched into registers; all 4 B fragments are loaded before the WMMAs.
// ---------------------------------------------------------------------------
template<int ACT>
__global__ __launch_bounds__(128)
void k_gemm_bf16(const __bf16* __restrict__ A, const __bf16* __restrict__ W,
                 const float* __restrict__ bias, float* __restrict__ Cout,
                 int M, int N, int Kp)
{
  __shared__ __bf16 Bs[2][64 * 32];
  const int lane = threadIdx.x & 31;
  const int wy = threadIdx.y;             // 0..3
  const int lo = lane & 15;
  const int hi = (lane >> 4) * 8;
  const int mt = blockIdx.y * 4 + wy;
  const int n0 = blockIdx.x * 64;
  const int m = mt * 16 + lo;
  const __bf16* arow = A + (size_t)imin(m, M - 1) * Kp;

#if USE_TDM
  const uint32_t bs_off0 = (uint32_t)(uintptr_t)&Bs[0][0];
  const uint32_t bs_off1 = (uint32_t)(uintptr_t)&Bs[1][0];
  const __bf16* wtile = W + (size_t)n0 * Kp;   // tile rows n0..n0+63 (padded alloc)
  if (wy == 0) {
    tdm_load_b_tile(wtile, bs_off0, Kp);
    __builtin_amdgcn_s_wait_tensorcnt(0);
  }
#else
  // cooperative B staging: thread tid owns row (tid>>1), 16-elem half (tid&1)
  const int tid = wy * 32 + lane;
  const int brow = tid >> 1;
  const int bhalf = (tid & 1) * 16;
  const __bf16* wrow = W + (size_t)imin(n0 + brow, N - 1) * Kp + bhalf;
  __bf16* const bdst0 = &Bs[0][brow * 32 + bhalf];
  __bf16* const bdst1 = &Bs[1][brow * 32 + bhalf];
  *(bf16x16*)bdst0 = *(const bf16x16*)wrow;
#endif
  bf16x16 ac = ldfrag(arow, 0, hi);
  __syncthreads();

  f32x8 acc[4] = {};
  int buf = 0;
  for (int kb = 0; kb < Kp; kb += 32) {
    const int kn = (kb + 32 < Kp) ? kb + 32 : kb;   // clamped: always valid
    __builtin_prefetch(arow + kb + 128, 0, 3);
    // fetch next: A to registers, B to the other LDS buffer
    const bf16x16 an = ldfrag(arow, kn, hi);
#if USE_TDM
    if (wy == 0) tdm_load_b_tile(wtile + kn, buf ? bs_off0 : bs_off1, Kp);
#else
    *(bf16x16*)(buf ? bdst0 : bdst1) = *(const bf16x16*)(wrow + kn);
#endif
    // consume current buffer: all fragments first, then the WMMAs
    const __bf16* bb = &Bs[buf][0];
    bf16x16 bq[4];
#pragma unroll
    for (int j = 0; j < 4; ++j) bq[j] = ldfrag_lds(bb + (j * 16 + lo) * 32, hi);
#pragma unroll
    for (int j = 0; j < 4; ++j)
      acc[j] = __builtin_amdgcn_wmma_f32_16x16x32_bf16(false, ac, false, bq[j],
                                                       (short)0, acc[j], false, false);
    ac = an;
    buf ^= 1;
#if USE_TDM
    if (wy == 0) __builtin_amdgcn_s_wait_tensorcnt(0);
#endif
    __syncthreads();
  }
#pragma unroll
  for (int j = 0; j < 4; ++j) {
    const int col = n0 + j * 16 + lo;
    if (col >= N) continue;
    const float bs = bias ? bias[col] : 0.f;
#pragma unroll
    for (int q = 0; q < 8; ++q) {
      const int row = mt * 16 + hi + q;
      if (row < M) Cout[(size_t)row * N + col] = act_apply<ACT>(acc[j][q] + bs);
    }
  }
}

// ---------------------------------------------------------------------------
// Implicit-GEMM 3x3 conv on zero-halo NHWC bf16 input (IMG, h+2, w+2, Cp),
// tap-major packed weights (Cout, 9*Cp).  4-wave / LDS-B / double-buffered;
// flattened (tap, cb) walk advances the input pointer incrementally.
// ---------------------------------------------------------------------------
template<int ACT>
__global__ __launch_bounds__(128)
void k_conv3_bf16(const __bf16* __restrict__ In, const __bf16* __restrict__ Wt,
                  const float* __restrict__ bias, float* __restrict__ Out,
                  int IMG, int h, int w, int Cp, int Cout)
{
  __shared__ __bf16 Bs[2][64 * 32];
  const int M = IMG * h * w;
  const int Kw = 9 * Cp;
  const int lane = threadIdx.x & 31;
  const int wy = threadIdx.y;
  const int tid = wy * 32 + lane;
  const int lo = lane & 15;
  const int hi = (lane >> 4) * 8;
  const int mt = blockIdx.y * 4 + wy;
  const int n0 = blockIdx.x * 64;
  const int mm = imin(mt * 16 + lo, M - 1);
  const int hw = h * w;
  const int img = mm / hw;
  const int p = mm % hw;
  const int py = p / w, px = p % w;
  // top-left of the 3x3 window in padded coords (tap 0)
  const __bf16* win = In + (((size_t)img * (h + 2) + py) * (w + 2) + px) * Cp;

  const int brow = tid >> 1;
  const int bhalf = (tid & 1) * 16;
  const __bf16* wrow = Wt + (size_t)imin(n0 + brow, Cout - 1) * Kw + bhalf;
  __bf16* const bdst0 = &Bs[0][brow * 32 + bhalf];
  __bf16* const bdst1 = &Bs[1][brow * 32 + bhalf];

  // stage block 0 (tap 0, cb 0)
  *(bf16x16*)bdst0 = *(const bf16x16*)wrow;
  bf16x16 ac = ldfrag(win, 0, hi);
  __syncthreads();

  f32x8 acc[4] = {};
  const int steps = 9 * (Cp >> 5);
  int buf = 0;
  int cb = 0, t3 = 0, wflat = 0;          // current block coords
  const __bf16* ap = win;                 // input base of current tap
  for (int s2 = 0; s2 < steps; ++s2) {
    // next block coords (clamped on last step -> redundant valid reload)
    int ncb = cb + 32;
    int nt3 = t3;
    const __bf16* nap = ap;
    if (ncb == Cp) { ncb = 0; nap += (size_t)(t3 == 2 ? w : 1) * Cp; nt3 = (t3 == 2) ? 0 : t3 + 1; }
    int nwflat = wflat + 32;
    if (s2 + 1 == steps) { ncb = cb; nap = ap; nt3 = t3; nwflat = wflat; }
    // fetch next
    const bf16x16 an = ldfrag(nap, ncb, hi);
    *(bf16x16*)(buf ? bdst0 : bdst1) = *(const bf16x16*)(wrow + nwflat);
    // consume current: fragments first, then WMMAs
    const __bf16* bb = &Bs[buf][0];
    bf16x16 bq[4];
#pragma unroll
    for (int j = 0; j < 4; ++j) bq[j] = ldfrag_lds(bb + (j * 16 + lo) * 32, hi);
#pragma unroll
    for (int j = 0; j < 4; ++j)
      acc[j] = __builtin_amdgcn_wmma_f32_16x16x32_bf16(false, ac, false, bq[j],
                                                       (short)0, acc[j], false, false);
    ac = an; ap = nap; cb = ncb; t3 = nt3; wflat = nwflat;
    buf ^= 1;
    __syncthreads();
  }
#pragma unroll
  for (int j = 0; j < 4; ++j) {
    const int col = n0 + j * 16 + lo;
    if (col >= Cout) continue;
    const float bs = bias[col];
#pragma unroll
    for (int q = 0; q < 8; ++q) {
      const int row = mt * 16 + hi + q;
      if (row < M) Out[(size_t)row * Cout + col] = act_apply<ACT>(acc[j][q] + bs);
    }
  }
}

// ---------------------------------------------------------------------------
// Pack kernels (fp32 -> padded bf16)
// ---------------------------------------------------------------------------
__global__ void k_pack_rows(const float* __restrict__ src, int lda,
                            __bf16* __restrict__ dst, int rows, int K, int Kp)
{
  const int total = rows * Kp;
  int idx = blockIdx.x * blockDim.x + threadIdx.x;
  if (idx >= total) return;
  const int k = idx % Kp;
  const int r = idx / Kp;
  dst[idx] = f2bf(k < K ? src[(size_t)r * lda + k] : 0.f);
}

// OIHW (Cout,Cin,3,3) -> tap-major bf16 (Cout, 9*Cp)
__global__ void k_pack_w3(const float* __restrict__ src, __bf16* __restrict__ dst,
                          int Cout, int Cin, int Cp)
{
  const int total = Cout * 9 * Cp;
  int idx = blockIdx.x * blockDim.x + threadIdx.x;
  if (idx >= total) return;
  const int ci = idx % Cp;
  const int rest = idx / Cp;
  const int tap = rest % 9;
  const int co = rest / 9;
  dst[idx] = f2bf(ci < Cin ? src[((size_t)co * Cin + ci) * 9 + tap] : 0.f);
}

// pixel-major fp32 (IMG*h*w, Cin) -> zero-halo NHWC bf16 (IMG, h+2, w+2, Cp)
__global__ void k_pack_halo(const float* __restrict__ src, __bf16* __restrict__ dst,
                            int IMG, int h, int w, int Cin, int Cp)
{
  const int total = IMG * (h + 2) * (w + 2) * Cp;
  int idx = blockIdx.x * blockDim.x + threadIdx.x;
  if (idx >= total) return;
  const int ci = idx % Cp;
  const int p2 = idx / Cp;
  const int x2 = p2 % (w + 2);
  const int q = p2 / (w + 2);
  const int y2 = q % (h + 2);
  const int img = q / (h + 2);
  const int y = y2 - 1, x = x2 - 1;
  float v = 0.f;
  if (ci < Cin && (unsigned)y < (unsigned)h && (unsigned)x < (unsigned)w)
    v = src[((size_t)img * h * w + (size_t)y * w + x) * Cin + ci];
  dst[idx] = f2bf(v);
}

// ---------------------------------------------------------------------------
// Elementwise / scan kernels (fp32 VALU)
// ---------------------------------------------------------------------------
__global__ void k_layernorm(const float* __restrict__ x, const float* __restrict__ g,
                            const float* __restrict__ b, float* __restrict__ y, int C)
{
  __shared__ float red[128];
  const int t = blockIdx.x;
  const float* xr = x + (size_t)t * C;
  float s = 0.f;
  for (int c = threadIdx.x; c < C; c += 128) s += xr[c];
  red[threadIdx.x] = s; __syncthreads();
  for (int o = 64; o > 0; o >>= 1) { if (threadIdx.x < o) red[threadIdx.x] += red[threadIdx.x + o]; __syncthreads(); }
  const float mean = red[0] / C;
  __syncthreads();
  float v = 0.f;
  for (int c = threadIdx.x; c < C; c += 128) { float d = xr[c] - mean; v += d * d; }
  red[threadIdx.x] = v; __syncthreads();
  for (int o = 64; o > 0; o >>= 1) { if (threadIdx.x < o) red[threadIdx.x] += red[threadIdx.x + o]; __syncthreads(); }
  const float rstd = rsqrtf(red[0] / C + 1e-5f);
  float* yr = y + (size_t)t * C;
  for (int c = threadIdx.x; c < C; c += 128) yr[c] = (xr[c] - mean) * rstd * g[c] + b[c];
}

// (N,C,V,H,W) -> tokens (T=N*L, C), L = V*H*W
__global__ void k_tok(const float* __restrict__ x5, float* __restrict__ xt,
                      int Nb, int C, int L)
{
  const int total = Nb * C * L;
  int idx = blockIdx.x * blockDim.x + threadIdx.x;
  if (idx >= total) return;
  const int c = idx % C;
  const int t = idx / C;
  const int l = t % L;
  const int n = t / L;
  xt[idx] = x5[((size_t)n * C + c) * L + l];
}

// tokens (T,C) -> (N,C,V,H,W)
__global__ void k_untok(const float* __restrict__ xt, float* __restrict__ x5,
                        int Nb, int C, int L)
{
  const int total = Nb * C * L;
  int idx = blockIdx.x * blockDim.x + threadIdx.x;
  if (idx >= total) return;
  const int l = idx % L;
  const int rest = idx / L;
  const int c = rest % C;
  const int n = rest / C;
  x5[idx] = xt[((size_t)n * L + l) * C + c];
}

// drop last channel: (N,C,L) -> (N,C-1,L)
__global__ void k_slice(const float* __restrict__ in5, float* __restrict__ out,
                        int Nb, int C, int L)
{
  const int Cm = C - 1;
  const int total = Nb * Cm * L;
  int idx = blockIdx.x * blockDim.x + threadIdx.x;
  if (idx >= total) return;
  const int l = idx % L;
  const int rest = idx / L;
  const int c = rest % Cm;
  const int n = rest / Cm;
  out[idx] = in5[((size_t)n * C + c) * L + l];
}

// causal depthwise conv (K=4) over x_in = xz[:, :d] + SiLU -> xc (T,d)
__global__ void k_dwconv_silu(const float* __restrict__ xz, const float* __restrict__ w,
                              const float* __restrict__ b, float* __restrict__ xc,
                              int Nb, int L, int d)
{
  const int total = Nb * L * d;
  int idx = blockIdx.x * blockDim.x + threadIdx.x;
  if (idx >= total) return;
  const int ch = idx % d;
  const int t = idx / d;
  const int l = t % L;
  float s = b[ch];
#pragma unroll
  for (int k = 0; k < 4; ++k) {
    const int li = l + k - 3;
    if (li >= 0) s += w[ch * 4 + k] * xz[(size_t)(t + k - 3) * (2 * d) + ch];
  }
  xc[idx] = s / (1.f + __expf(-s));   // silu
}

// selective scan: one thread per (batch, channel); s=16 states in registers
__global__ void k_scan(const float* __restrict__ dt, const float* __restrict__ xc,
                       const float* __restrict__ dbl, const float* __restrict__ xz,
                       const float* __restrict__ A_log, const float* __restrict__ Dp,
                       float* __restrict__ y, int Nb, int L, int d, int r)
{
  int idx = blockIdx.x * blockDim.x + threadIdx.x;
  if (idx >= Nb * d) return;
  const int n = idx / d, ch = idx % d;
  const int cols = r + 32;
  float a[16], h[16];
#pragma unroll
  for (int j = 0; j < 16; ++j) { a[j] = -__expf(A_log[ch * 16 + j]); h[j] = 0.f; }
  const float Dch = Dp[ch];
  for (int l = 0; l < L; ++l) {
    const size_t t = (size_t)n * L + l;
    const float dtv = dt[t * d + ch];
    const float xv  = xc[t * d + ch];
    const float dx  = dtv * xv;
    const float* bc = dbl + t * cols + r;   // [Bm(16) | Cm(16)]
    float acc = 0.f;
#pragma unroll
    for (int j = 0; j < 16; ++j) {
      const float hj = __expf(dtv * a[j]) * h[j] + dx * bc[j];
      h[j] = hj;
      acc += hj * bc[16 + j];
    }
    const float zv = xz[t * (2 * (size_t)d) + d + ch];
    y[t * d + ch] = (acc + xv * Dch) * (zv / (1.f + __expf(-zv)));
  }
}

// prev5 (N,c0,V,h0,w0) bilinear(align-corners) + prj5 (N,c2,V,h,w) -> NHWC (N*V,h,w,c0+c2)
__global__ void k_up_concat(const float* __restrict__ prev5, const float* __restrict__ prj5,
                            float* __restrict__ out,
                            int Nb, int V, int c0, int c2, int h0, int w0, int h, int w)
{
  const int Ct = c0 + c2;
  const int total = Nb * V * h * w * Ct;
  int idx = blockIdx.x * blockDim.x + threadIdx.x;
  if (idx >= total) return;
  const int c = idx % Ct;
  const int p = idx / Ct;
  const int x = p % w;
  const int q = p / w;
  const int yy = q % h;
  const int img = q / h;
  const int v = img % V, n = img / V;
  float val;
  if (c < c0) {
    const float sy = (float)(h0 - 1) / (float)(h - 1);
    const float sx = (float)(w0 - 1) / (float)(w - 1);
    const float fy = yy * sy, fx = x * sx;
    const int y0 = (int)fy, x0 = (int)fx;
    const int y1 = y0 + 1 < h0 ? y0 + 1 : h0 - 1;
    const int x1 = x0 + 1 < w0 ? x0 + 1 : w0 - 1;
    const float wy = fy - y0, wx = fx - x0;
    const float* bp = prev5 + (((size_t)n * c0 + c) * V + v) * (size_t)(h0 * w0);
    const float r0 = bp[y0 * w0 + x0] * (1.f - wy) + bp[y1 * w0 + x0] * wy;
    const float r1 = bp[y0 * w0 + x1] * (1.f - wy) + bp[y1 * w0 + x1] * wy;
    val = r0 * (1.f - wx) + r1 * wx;
  } else {
    val = prj5[(((size_t)n * c2 + (c - c0)) * V + v) * (size_t)(h * w) + yy * w + x];
  }
  out[idx] = val;
}

// faithful .view gather: src flat (N, C*V, HW) reinterpreted (N,V,C,HW)
__global__ void k_mha_gather(const float* __restrict__ src, float* __restrict__ A,
                             int Nb, int HW, int C, int V)
{
  const int total = V * Nb * HW * C;
  int idx = blockIdx.x * blockDim.x + threadIdx.x;
  if (idx >= total) return;
  const int c = idx % C;
  int row = idx / C;
  const int p = row % HW;
  row /= HW;
  const int n = row % Nb;
  const int v = row / Nb;
  A[idx] = src[(((size_t)n * V + v) * C + c) * HW + p];
}

// 2-view attention: qkv rows (v*NHW + b), cols [q|k|v] of C each
__global__ void k_attn2(const float* __restrict__ qkv, float* __restrict__ o,
                        int NHW, int C, int heads)
{
  int idx = blockIdx.x * blockDim.x + threadIdx.x;
  if (idx >= NHW * heads) return;
  const int hh = idx % heads;
  const int bk = idx / heads;
  const int dh = C / heads;
  const float* q0 = qkv + (size_t)bk * (3 * C) + hh * dh;
  const float* q1 = qkv + ((size_t)NHW + bk) * (3 * C) + hh * dh;
  const float* k0 = q0 + C; const float* k1 = q1 + C;
  const float* v0 = k0 + C; const float* v1 = k1 + C;
  float s00 = 0, s01 = 0, s10 = 0, s11 = 0;
  for (int d2 = 0; d2 < dh; ++d2) {
    const float a0 = q0[d2], a1 = q1[d2], b0 = k0[d2], b1 = k1[d2];
    s00 += a0 * b0; s01 += a0 * b1; s10 += a1 * b0; s11 += a1 * b1;
  }
  const float sc = rsqrtf((float)dh);
  s00 *= sc; s01 *= sc; s10 *= sc; s11 *= sc;
  float m0 = s00 > s01 ? s00 : s01;
  float e00 = __expf(s00 - m0), e01 = __expf(s01 - m0);
  float i0 = 1.f / (e00 + e01);
  float m1 = s10 > s11 ? s10 : s11;
  float e10 = __expf(s10 - m1), e11 = __expf(s11 - m1);
  float i1 = 1.f / (e10 + e11);
  float* o0 = o + (size_t)bk * C + hh * dh;
  float* o1 = o + ((size_t)NHW + bk) * C + hh * dh;
  for (int d2 = 0; d2 < dh; ++d2) {
    o0[d2] = (e00 * i0) * v0[d2] + (e01 * i0) * v1[d2];
    o1[d2] = (e10 * i1) * v0[d2] + (e11 * i1) * v1[d2];
  }
}

__global__ void k_mean2(const float* __restrict__ a, float* __restrict__ out, int n)
{
  int i = blockIdx.x * blockDim.x + threadIdx.x;
  if (i < n) out[i] = 0.5f * (a[i] + a[(size_t)n + i]);
}

// ---------------------------------------------------------------------------
// Host orchestration
// ---------------------------------------------------------------------------
static inline int cd(long a, int b) { return (int)((a + b - 1) / b); }

struct Arena { char* base; size_t off; size_t reset_pt; };
static void* ar_alloc_b(Arena& a, size_t bytes) {
  void* p = a.base + a.off;
  a.off += (bytes + 255) & ~(size_t)255;
  return p;
}
static float*  ar_f(Arena& a, size_t n) { return (float*)ar_alloc_b(a, n * 4); }
static __bf16* ar_h(Arena& a, size_t n) { return (__bf16*)ar_alloc_b(a, n * 2); }
static void ar_reset(Arena& a) { a.off = a.reset_pt; }

struct MP { const float *A_log, *Dp, *conv_b, *conv_w, *dt_b, *dt_w, *in_w, *ln_b, *ln_g, *out_w, *x_w; };
struct FP { const float *b1, *b2, *b3, *w1, *w2, *w3; };

// pack A (M,K stride lda) and W (N,K) to bf16/Kp, then WMMA GEMM.
// W buffer is padded by 64 extra rows so TDM edge-tile DMA stays in-bounds.
static void gemm_bf16(hipStream_t st, Arena& ar, const float* A, int lda,
                      const float* W, const float* bias, float* C,
                      int M, int N, int K, int act)
{
  const int Kp = (K + 31) & ~31;
  __bf16* Ab = ar_h(ar, (size_t)M * Kp);
  __bf16* Wb = ar_h(ar, ((size_t)N + 64) * Kp);
  k_pack_rows<<<cd((long)M * Kp, 256), 256, 0, st>>>(A, lda, Ab, M, K, Kp);
  k_pack_rows<<<cd((long)N * Kp, 256), 256, 0, st>>>(W, K, Wb, N, K, Kp);
  dim3 g(cd(N, 64), cd(M, 64)), b(32, 4);
  switch (act) {
    case 1:  k_gemm_bf16<1><<<g, b, 0, st>>>(Ab, Wb, bias, C, M, N, Kp); break;
    case 2:  k_gemm_bf16<2><<<g, b, 0, st>>>(Ab, Wb, bias, C, M, N, Kp); break;
    case 3:  k_gemm_bf16<3><<<g, b, 0, st>>>(Ab, Wb, bias, C, M, N, Kp); break;
    default: k_gemm_bf16<0><<<g, b, 0, st>>>(Ab, Wb, bias, C, M, N, Kp); break;
  }
}

// pixel-major fp32 input (IMG*h*w, Cin) -> halo-packed implicit-GEMM conv3x3
static void conv3_bf16(hipStream_t st, Arena& ar, const float* In, const float* W,
                       const float* bias, float* Out,
                       int IMG, int h, int w, int Cin, int Cout, int act)
{
  const int Cp = (Cin + 31) & ~31;
  __bf16* Hb = ar_h(ar, (size_t)IMG * (h + 2) * (w + 2) * Cp);
  __bf16* Wb = ar_h(ar, (size_t)Cout * 9 * Cp);
  k_pack_halo<<<cd((long)IMG * (h + 2) * (w + 2) * Cp, 256), 256, 0, st>>>(In, Hb, IMG, h, w, Cin, Cp);
  k_pack_w3<<<cd((long)Cout * 9 * Cp, 256), 256, 0, st>>>(W, Wb, Cout, Cin, Cp);
  dim3 g(cd(Cout, 64), cd((long)IMG * h * w, 64)), b(32, 4);
  if (act == 1) k_conv3_bf16<1><<<g, b, 0, st>>>(Hb, Wb, bias, Out, IMG, h, w, Cp, Cout);
  else          k_conv3_bf16<0><<<g, b, 0, st>>>(Hb, Wb, bias, Out, IMG, h, w, Cp, Cout);
}

// xt: (T,dim) tokens -> out5: (N,dim,V,H,W)
static void run_mamba(hipStream_t st, Arena& ar, const MP& P,
                      const float* xt, float* out5, int Nb, int dim, int L)
{
  const int T = Nb * L, d = 2 * dim, r = (dim + 15) / 16, cols = r + 32;
  float* xn = ar_f(ar, (size_t)T * dim);
  k_layernorm<<<T, 128, 0, st>>>(xt, P.ln_g, P.ln_b, xn, dim);
  float* xz = ar_f(ar, (size_t)T * 2 * d);
  gemm_bf16(st, ar, xn, dim, P.in_w, nullptr, xz, T, 2 * d, dim, 0);
  float* xc = ar_f(ar, (size_t)T * d);
  k_dwconv_silu<<<cd((long)T * d, 256), 256, 0, st>>>(xz, P.conv_w, P.conv_b, xc, Nb, L, d);
  float* dbl = ar_f(ar, (size_t)T * cols);
  gemm_bf16(st, ar, xc, d, P.x_w, nullptr, dbl, T, cols, d, 0);
  float* dt = ar_f(ar, (size_t)T * d);
  gemm_bf16(st, ar, dbl, cols, P.dt_w, P.dt_b, dt, T, d, r, 3 /*softplus*/);
  float* y = ar_f(ar, (size_t)T * d);
  k_scan<<<cd((long)Nb * d, 128), 128, 0, st>>>(dt, xc, dbl, xz, P.A_log, P.Dp, y, Nb, L, d, r);
  float* otc = ar_f(ar, (size_t)T * dim);
  gemm_bf16(st, ar, y, d, P.out_w, nullptr, otc, T, dim, d, 0);
  k_untok<<<cd((long)T * dim, 256), 256, 0, st>>>(otc, out5, Nb, dim, L);
}

// fused output is pixel-major (IMG*hw, o) == Mamba token layout (T, o)
static void run_fuse(hipStream_t st, Arena& ar, const FP& P,
                     const float* prev5, const float* prj5, float* fused,
                     int Nb, int V, int c0, int c2, int o,
                     int h0, int w0, int h, int w)
{
  const int IMG = Nb * V, hw = h * w, M = IMG * hw, Ct = c0 + c2, C4 = 4 * o;
  float* mf = ar_f(ar, (size_t)M * Ct);
  k_up_concat<<<cd((long)M * Ct, 256), 256, 0, st>>>(prev5, prj5, mf, Nb, V, c0, c2, h0, w0, h, w);
  float* c1 = ar_f(ar, (size_t)M * C4);
  gemm_bf16(st, ar, mf, Ct, P.w1, P.b1, c1, M, C4, Ct, 1 /*relu*/);    // 1x1 conv
  float* c2b = ar_f(ar, (size_t)M * C4);
  conv3_bf16(st, ar, c1, P.w2, P.b2, c2b, IMG, h, w, C4, C4, 1);
  conv3_bf16(st, ar, c2b, P.w3, P.b3, fused, IMG, h, w, C4, o, 0);
}

extern "C" void kernel_launch(void* const* d_in, const int* in_sizes, int n_in,
                              void* d_out, int out_size, void* d_ws, size_t ws_size,
                              hipStream_t stream)
{
  auto F = [&](int i) { return (const float*)d_in[i]; };
  const float* prj1 = F(0);
  const float* prj2 = F(1);
  const float* prj3 = F(2);
  const float* prj4 = F(3);
  // jax pytree order: dict keys sorted. Mamba: A_log,D,conv_b,conv_w,dt_b,dt_w,in_w,ln_b,ln_g,out_w,x_w
  auto mp = [&](int b) { MP p; p.A_log = F(b); p.Dp = F(b + 1); p.conv_b = F(b + 2);
    p.conv_w = F(b + 3); p.dt_b = F(b + 4); p.dt_w = F(b + 5); p.in_w = F(b + 6);
    p.ln_b = F(b + 7); p.ln_g = F(b + 8); p.out_w = F(b + 9); p.x_w = F(b + 10); return p; };
  auto fp = [&](int b) { FP p; p.b1 = F(b); p.b2 = F(b + 1); p.b3 = F(b + 2);
    p.w1 = F(b + 3); p.w2 = F(b + 4); p.w3 = F(b + 5); return p; };
  const MP enc1 = mp(4), enc2 = mp(15), enc3 = mp(26), enc4 = mp(37);
  const FP fuse2 = fp(48), fuse3 = fp(54), fuse4 = fp(60);
  const float* mha_in_b  = F(66);
  const float* mha_in_w  = F(67);
  const float* mha_out_b = F(68);
  const float* mha_out_w = F(69);

  // workspace: 3 persistent 1M-float slots + arena
  char* wsb = (char*)d_ws;
  const size_t SLOT = (size_t)1024 * 1024 * sizeof(float);
  float* P0 = (float*)(wsb);              // enc4 out / enc2 out
  float* P1 = (float*)(wsb + SLOT);       // enc3 out / enc1 out
  float* P2 = (float*)(wsb + 2 * SLOT);   // fused tokens
  Arena ar; ar.base = wsb; ar.reset_pt = 3 * SLOT; ar.off = ar.reset_pt;

  // output layout: vout | enc3[:,:-1] | enc2[:,:-1] | enc1[:,:-1]
  float* out_v  = (float*)d_out;          // 2*96*48*48   = 442368
  float* out_f0 = out_v  + 442368;        // 2*384*2*12*12= 221184
  float* out_f1 = out_f0 + 221184;        // 2*192*2*24*24= 442368
  float* out_f2 = out_f1 + 442368;        // 2*96*2*48*48 = 884736

  const int Nb = 2, V = 2;

  // ---- enc4 on prj4 (dim=769, 6x6) -> P0
  {
    const int dim = 769, L = V * 6 * 6, T = Nb * L;
    ar_reset(ar);
    float* xt = ar_f(ar, (size_t)T * dim);
    k_tok<<<cd((long)T * dim, 256), 256, 0, stream>>>(prj4, xt, Nb, dim, L);
    run_mamba(stream, ar, enc4, xt, P0, Nb, dim, L);
  }
  // ---- level 3: fuse4(prev=P0 6x6 c769, prj3 c385) -> enc3 (dim=385, 12x12) -> P1
  {
    const int c0 = 769, c2 = 385, o = 385, h0 = 6, w0 = 6, h = 12, w = 12;
    const int L = V * h * w;
    ar_reset(ar);
    run_fuse(stream, ar, fuse4, P0, prj3, P2, Nb, V, c0, c2, o, h0, w0, h, w);
    ar_reset(ar);
    run_mamba(stream, ar, enc3, P2, P1, Nb, o, L);
    k_slice<<<cd((long)Nb * (o - 1) * L, 256), 256, 0, stream>>>(P1, out_f0, Nb, o, L);
  }
  // ---- level 2: fuse3(prev=P1 12x12 c385, prj2 c193) -> enc2 (dim=193, 24x24) -> P0
  {
    const int c0 = 385, c2 = 193, o = 193, h0 = 12, w0 = 12, h = 24, w = 24;
    const int L = V * h * w;
    ar_reset(ar);
    run_fuse(stream, ar, fuse3, P1, prj2, P2, Nb, V, c0, c2, o, h0, w0, h, w);
    ar_reset(ar);
    run_mamba(stream, ar, enc2, P2, P0, Nb, o, L);
    k_slice<<<cd((long)Nb * (o - 1) * L, 256), 256, 0, stream>>>(P0, out_f1, Nb, o, L);
  }
  // ---- level 1: fuse2(prev=P0 24x24 c193, prj1 c97) -> enc1 (dim=97, 48x48) -> P1
  {
    const int c0 = 193, c2 = 97, o = 97, h0 = 24, w0 = 24, h = 48, w = 48;
    const int L = V * h * w;
    ar_reset(ar);
    run_fuse(stream, ar, fuse2, P0, prj1, P2, Nb, V, c0, c2, o, h0, w0, h, w);
    ar_reset(ar);
    run_mamba(stream, ar, enc1, P2, P1, Nb, o, L);
    k_slice<<<cd((long)Nb * (o - 1) * L, 256), 256, 0, stream>>>(P1, out_f2, Nb, o, L);
  }
  // ---- MHA view fuse on enc1[:, :-1] (C=96) which already sits in out_f2
  {
    const int C = 96, HW = 48 * 48, NHW = Nb * HW, Rows = V * NHW, heads = 8;
    ar_reset(ar);
    float* Ab = ar_f(ar, (size_t)Rows * C);
    k_mha_gather<<<cd((long)Rows * C, 256), 256, 0, stream>>>(out_f2, Ab, Nb, HW, C, V);
    float* qkv = ar_f(ar, (size_t)Rows * 3 * C);
    gemm_bf16(stream, ar, Ab, C, mha_in_w, mha_in_b, qkv, Rows, 3 * C, C, 0);
    float* ob = ar_f(ar, (size_t)Rows * C);
    k_attn2<<<cd((long)NHW * heads, 256), 256, 0, stream>>>(qkv, ob, NHW, C, heads);
    float* op = ar_f(ar, (size_t)Rows * C);
    gemm_bf16(stream, ar, ob, C, mha_out_w, mha_out_b, op, Rows, C, C, 0);
    k_mean2<<<cd((long)NHW * C, 256), 256, 0, stream>>>(op, out_v, NHW * C);
  }
  (void)in_sizes; (void)n_in; (void)out_size; (void)ws_size;
}
